// TreeGPT_10282151707529
// MI455X (gfx1250) — matrix operand order repaired and checked by reading, not compiled, not run on hardware
//
#include <hip/hip_runtime.h>
#include <math.h>

typedef __attribute__((ext_vector_type(16))) _Float16 v16h;
typedef __attribute__((ext_vector_type(8)))  _Float16 v8h;
typedef __attribute__((ext_vector_type(4)))  _Float16 v4h;
typedef __attribute__((ext_vector_type(8)))  float    v8f;

#define TPB 256
#define BM  64
#define BN  128
#define BK  32

#define SHUF16(lo, hi) __builtin_shufflevector(lo, hi, 0,1,2,3,4,5,6,7,8,9,10,11,12,13,14,15)

// ---------------------------------------------------------------------------
// Stage one 64x32 A tile and one 32x128 B tile (converted f32->f16) into LDS.
// Branch-free: unconditional float4 loads with clamped addresses + selects.
// ---------------------------------------------------------------------------
template<bool TRB, bool EDGE>
__device__ __forceinline__ void stage_tiles(
    _Float16 (* __restrict__ As)[40], _Float16 (* __restrict__ Bt)[40],
    const float* __restrict__ A, long long lda,
    const float* __restrict__ Bw, long long ldb,
    int m0, int n0, int kb, int M, int N, int Sedge, int tid)
{
    #pragma unroll
    for (int i = 0; i < 2; ++i) {
        int c  = tid + i * TPB;           // 0..511
        int r  = c >> 3;                  // 0..63
        int kq = (c & 7) * 4;             // 0..28
        int grow = m0 + r;
        bool ok = grow < M;
        int gr = ok ? grow : (M - 1);
        long long ar = gr;
        if (EDGE)
            ar = (long long)(gr / (Sedge - 1)) * Sedge + (gr % (Sedge - 1));
        float4 va = *(const float4*)(A + ar * lda + (kb + kq));
        v4h hv;
        hv[0] = (_Float16)(ok ? va.x : 0.f);
        hv[1] = (_Float16)(ok ? va.y : 0.f);
        hv[2] = (_Float16)(ok ? va.z : 0.f);
        hv[3] = (_Float16)(ok ? va.w : 0.f);
        *(v4h*)&As[r][kq] = hv;
    }
    if (TRB) {
        #pragma unroll
        for (int i = 0; i < 4; ++i) {
            int c  = tid + i * TPB;       // 0..1023
            int nn = c >> 3;              // 0..127
            int kq = (c & 7) * 4;         // 0..28
            int gn = n0 + nn;
            bool ok = gn < N;
            int gns = ok ? gn : (N - 1);
            float4 vb = *(const float4*)(Bw + (long long)gns * ldb + (kb + kq));
            v4h hv;
            hv[0] = (_Float16)(ok ? vb.x : 0.f);
            hv[1] = (_Float16)(ok ? vb.y : 0.f);
            hv[2] = (_Float16)(ok ? vb.z : 0.f);
            hv[3] = (_Float16)(ok ? vb.w : 0.f);
            *(v4h*)&Bt[nn][kq] = hv;
        }
    } else {
        #pragma unroll
        for (int i = 0; i < 4; ++i) {
            int c  = tid + i * TPB;       // 0..1023
            int kk = c >> 5;              // 0..31
            int nq = (c & 31) * 4;        // 0..124
            int gn = n0 + nq;
            bool ok = gn < N;             // N is a multiple of 4 everywhere
            int gns = ok ? gn : (N - 4);
            float4 vb = *(const float4*)(Bw + (long long)(kb + kk) * ldb + gns);
            Bt[nq + 0][kk] = (_Float16)(ok ? vb.x : 0.f);
            Bt[nq + 1][kk] = (_Float16)(ok ? vb.y : 0.f);
            Bt[nq + 2][kk] = (_Float16)(ok ? vb.z : 0.f);
            Bt[nq + 3][kk] = (_Float16)(ok ? vb.w : 0.f);
        }
    }
}

// ---------------------------------------------------------------------------
// f32-in / f32-out GEMM on the WMMA f16 path, fp32 accumulate.
//   C[m,n] = epilogue( sum_k A[m,k] * B(k,n) )
// B(k,n) = TRB ? Bw[n*ldb + k] : Bw[k*ldb + n]
// MODE: 0 plain | 1 +bias | 2 +bias+res | 3 +bias,relu | 4 res + sigmoid(acc+bias)*mul
// EDGE: A row m -> (m/(S-1))*S + m%(S-1); C row m -> same + 1  (TreeFFN shift)
// Batch z: base += (z/Hdiv)*s1 + (z%Hdiv)*s2  (head-strided batching)
// Double-buffered k-loop: one barrier per k-step; staging of tile i+1
// overlaps the WMMAs of tile i.
// ---------------------------------------------------------------------------
template<int MODE, bool TRB, bool EDGE>
__global__ __launch_bounds__(TPB) void gemm_wmma_kernel(
    const float* __restrict__ A, long long lda,
    const float* __restrict__ Bw, long long ldb,
    const float* __restrict__ bias,
    const float* __restrict__ res,
    const float* __restrict__ mul,
    float* __restrict__ C, long long ldc,
    int M, int N, int K, int Sedge, int Hdiv,
    long long sA1, long long sA2,
    long long sB1, long long sB2,
    long long sC1, long long sC2)
{
    __shared__ _Float16 As[2][BM][40];   // [buf][row][k]
    __shared__ _Float16 Bt[2][BN][40];   // [buf][col][k] (transposed)

    const int tid   = threadIdx.x;
    const int lane  = tid & 31;
    const int wave  = tid >> 5;
    const int waveM = wave >> 2;          // 0..1 : 32 rows each
    const int waveN = wave & 3;           // 0..3 : 32 cols each
    const int half  = lane >> 4;
    const int l16   = lane & 15;

    {
        const int z = blockIdx.z;
        const long long zq = z / Hdiv, zr = z % Hdiv;
        A  += zq * sA1 + zr * sA2;
        Bw += zq * sB1 + zr * sB2;
        C  += zq * sC1 + zr * sC2;
    }

    const int m0 = blockIdx.y * BM;
    const int n0 = blockIdx.x * BN;

    v8f acc00 = {}, acc01 = {}, acc10 = {}, acc11 = {};

    stage_tiles<TRB, EDGE>(As[0], Bt[0], A, lda, Bw, ldb,
                           m0, n0, 0, M, N, Sedge, tid);
    __syncthreads();

    int buf = 0;
    for (int kb = 0; kb < K; kb += BK, buf ^= 1) {
        // ---- fragments from current buffer: two ds_load_b128 each ----
        const int ra0 = waveM * 32 + l16;
        const int ra1 = ra0 + 16;
        v8h lo, hi;
        lo = *(const v8h*)&As[buf][ra0][half * 8];
        hi = *(const v8h*)&As[buf][ra0][16 + half * 8];
        v16h a0 = SHUF16(lo, hi);
        lo = *(const v8h*)&As[buf][ra1][half * 8];
        hi = *(const v8h*)&As[buf][ra1][16 + half * 8];
        v16h a1 = SHUF16(lo, hi);

        const int cb0 = waveN * 32 + l16;
        lo = *(const v8h*)&Bt[buf][cb0][half * 8];
        hi = *(const v8h*)&Bt[buf][cb0][16 + half * 8];
        v16h b0 = SHUF16(lo, hi);
        lo = *(const v8h*)&Bt[buf][cb0 + 16][half * 8];
        hi = *(const v8h*)&Bt[buf][cb0 + 16][16 + half * 8];
        v16h b1 = SHUF16(lo, hi);

        // ---- prefetch next tile into the other buffer (no conflict) ----
        if (kb + BK < K)
            stage_tiles<TRB, EDGE>(As[buf ^ 1], Bt[buf ^ 1], A, lda, Bw, ldb,
                                   m0, n0, kb + BK, M, N, Sedge, tid);

        acc00 = __builtin_amdgcn_wmma_f32_16x16x32_f16(false, a0, false, b0, (short)0, acc00, false, false);
        acc01 = __builtin_amdgcn_wmma_f32_16x16x32_f16(false, a0, false, b1, (short)0, acc01, false, false);
        acc10 = __builtin_amdgcn_wmma_f32_16x16x32_f16(false, a1, false, b0, (short)0, acc10, false, false);
        acc11 = __builtin_amdgcn_wmma_f32_16x16x32_f16(false, a1, false, b1, (short)0, acc11, false, false);

        __syncthreads();
    }

    // ---- epilogue: C frag layout -> M = r + 8*half, N = lane&15 ----
    const bool full = (m0 + BM <= M) && (n0 + BN <= N);
    const int colBase = n0 + waveN * 32 + l16;
    const int rowBase = m0 + waveM * 32 + 8 * half;

    #pragma unroll
    for (int mt = 0; mt < 2; ++mt) {
        #pragma unroll
        for (int nt = 0; nt < 2; ++nt) {
            v8f cc = (mt == 0) ? (nt == 0 ? acc00 : acc01)
                               : (nt == 0 ? acc10 : acc11);
            const int col  = colBase + nt * 16;
            const int row0 = rowBase + mt * 16;

            if (full) {
                float bv_ = 0.f;
                if (MODE >= 1) bv_ = bias[col];
                long long base = (long long)row0 * ldc + col;
                const float* rp = res + base;      // valid only for MODE 2/4
                const float* mp = mul + base;      // valid only for MODE 4
                #pragma unroll
                for (int r = 0; r < 8; ++r) {
                    float a_ = cc[r] + bv_;
                    float out;
                    if (MODE == 3)      out = a_ > 0.f ? a_ : 0.f;
                    else if (MODE == 4) {
                        float g = 1.f / (1.f + __expf(-a_));
                        out = *rp + g * (*mp);
                    }
                    else if (MODE == 2) out = a_ + *rp;
                    else                out = a_;

                    long long coff = base;
                    if (EDGE) {
                        int row = row0 + r;
                        long long crow = (long long)(row / (Sedge - 1)) * Sedge
                                         + (row % (Sedge - 1)) + 1;
                        coff = crow * ldc + col;
                    }
                    C[coff] = out;
                    base += ldc;
                    rp += ldc;
                    mp += ldc;
                }
            } else {
                float bv_ = 0.f;
                if (MODE >= 1 && col < N) bv_ = bias[col];
                #pragma unroll
                for (int r = 0; r < 8; ++r) {
                    int row = row0 + r;
                    if (row < M && col < N) {
                        float a_ = cc[r] + bv_;
                        float out;
                        long long ix = (long long)row * ldc + col;
                        if (MODE == 3)      out = a_ > 0.f ? a_ : 0.f;
                        else if (MODE == 4) {
                            float g = 1.f / (1.f + __expf(-a_));
                            out = res[ix] + g * mul[ix];
                        }
                        else if (MODE == 2) out = a_ + res[ix];
                        else                out = a_;

                        long long crow = row;
                        if (EDGE)
                            crow = (long long)(row / (Sedge - 1)) * Sedge
                                   + (row % (Sedge - 1)) + 1;
                        C[crow * ldc + col] = out;
                    }
                }
            }
        }
    }
}

template<int MODE, bool TRB, bool EDGE>
static void gemm_go(const float* A, long long lda, const float* Bw, long long ldb,
                    const float* bias, const float* res, const float* mul,
                    float* C, long long ldc, int M, int N, int K,
                    int Sedge, int Z, int Hdiv,
                    long long sA1, long long sA2,
                    long long sB1, long long sB2,
                    long long sC1, long long sC2, hipStream_t stream)
{
    dim3 g((N + BN - 1) / BN, (M + BM - 1) / BM, Z);
    gemm_wmma_kernel<MODE, TRB, EDGE><<<g, dim3(TPB), 0, stream>>>(
        A, lda, Bw, ldb, bias, res, mul, C, ldc, M, N, K, Sedge, Hdiv,
        sA1, sA2, sB1, sB2, sC1, sC2);
}

// ---------------------------------------------------------------------------
__global__ __launch_bounds__(TPB) void embed_kernel(
    const int* __restrict__ ids, const float* __restrict__ tok,
    const float* __restrict__ pos, float* __restrict__ h,
    int S, int D, long long total)
{
    long long i = (long long)blockIdx.x * TPB + threadIdx.x;
    if (i >= total) return;
    int d = (int)(i % D);
    long long bs = i / D;
    int s = (int)(bs % S);
    int id = ids[bs];
    h[i] = tok[(long long)id * D + d] + pos[(long long)s * D + d];
}

__global__ __launch_bounds__(TPB) void ln_kernel(
    const float* __restrict__ in, const float* __restrict__ w,
    const float* __restrict__ b, float* __restrict__ out, int D)
{
    __shared__ float red[TPB];
    const int tid = threadIdx.x;
    const float* x = in + (long long)blockIdx.x * D;

    float s = 0.f;
    for (int i = tid; i < D; i += TPB) s += x[i];
    red[tid] = s; __syncthreads();
    for (int off = TPB / 2; off > 0; off >>= 1) {
        if (tid < off) red[tid] += red[tid + off];
        __syncthreads();
    }
    float mu = red[0] / D; __syncthreads();

    float v = 0.f;
    for (int i = tid; i < D; i += TPB) { float d = x[i] - mu; v += d * d; }
    red[tid] = v; __syncthreads();
    for (int off = TPB / 2; off > 0; off >>= 1) {
        if (tid < off) red[tid] += red[tid + off];
        __syncthreads();
    }
    float rstd = rsqrtf(red[0] / D + 1e-5f);

    float* o = out + (long long)blockIdx.x * D;
    for (int i = tid; i < D; i += TPB)
        o[i] = (x[i] - mu) * rstd * w[i] + b[i];
}

__global__ __launch_bounds__(TPB) void softmax_causal_kernel(
    float* __restrict__ sc, int S, float scale)
{
    __shared__ float red[TPB];
    const int tid = threadIdx.x;
    long long row = blockIdx.x;
    int qi = (int)(row % S);
    float* p = sc + row * (long long)S;

    float mx = -3.0e38f;
    for (int j = tid; j < S; j += TPB) {
        float v = (j <= qi) ? p[j] * scale : -3.0e38f;
        mx = fmaxf(mx, v);
    }
    red[tid] = mx; __syncthreads();
    for (int off = TPB / 2; off > 0; off >>= 1) {
        if (tid < off) red[tid] = fmaxf(red[tid], red[tid + off]);
        __syncthreads();
    }
    mx = red[0]; __syncthreads();

    float sum = 0.f;
    for (int j = tid; j < S; j += TPB)
        if (j <= qi) sum += __expf(p[j] * scale - mx);
    red[tid] = sum; __syncthreads();
    for (int off = TPB / 2; off > 0; off >>= 1) {
        if (tid < off) red[tid] += red[tid + off];
        __syncthreads();
    }
    float inv = 1.f / red[0];

    for (int j = tid; j < S; j += TPB)
        p[j] = (j <= qi) ? __expf(p[j] * scale - mx) * inv : 0.f;
}

__global__ __launch_bounds__(TPB) void add_inplace_kernel(
    float* __restrict__ a, const float* __restrict__ b, long long n)
{
    long long i = (long long)blockIdx.x * TPB + threadIdx.x;
    if (i < n) a[i] += b[i];
}

__global__ __launch_bounds__(TPB) void zero_first_rows_kernel(
    float* __restrict__ agg, int S, int D)
{
    long long base = (long long)blockIdx.x * S * D;
    for (int d = threadIdx.x; d < D; d += TPB) agg[base + d] = 0.f;
}

// ---------------------------------------------------------------------------
extern "C" void kernel_launch(void* const* d_in, const int* in_sizes, int n_in,
                              void* d_out, int out_size, void* d_ws, size_t ws_size,
                              hipStream_t stream)
{
    const int*   ids   = (const int*)  d_in[0];
    const float* tok   = (const float*)d_in[1];
    const float* pos   = (const float*)d_in[2];
    const float* Wq    = (const float*)d_in[3];
    const float* bq    = (const float*)d_in[4];
    const float* Wk    = (const float*)d_in[5];
    const float* bk    = (const float*)d_in[6];
    const float* Wv    = (const float*)d_in[7];
    const float* bv    = (const float*)d_in[8];
    const float* Wo    = (const float*)d_in[9];
    const float* bo    = (const float*)d_in[10];
    const float* ln1w  = (const float*)d_in[11];
    const float* ln1b  = (const float*)d_in[12];
    const float* ln2w  = (const float*)d_in[13];
    const float* ln2b  = (const float*)d_in[14];
    const float* Win   = (const float*)d_in[15];
    const float* bin_  = (const float*)d_in[16];
    const float* Wedge = (const float*)d_in[17];
    const float* bedge = (const float*)d_in[18];
    const float* Wgate = (const float*)d_in[19];
    const float* bgate = (const float*)d_in[20];
    const float* lnfw  = (const float*)d_in[21];
    const float* lnfb  = (const float*)d_in[22];
    const float* Whead = (const float*)d_in[23];
    float* out = (float*)d_out;

    const int Bn = 4, Sn = 1024, Dn = 512, Hn = 8, Ln = 6, Vn = 32000, DHn = 64;
    const int M = Bn * Sn;                      // 4096
    const long long MD = (long long)M * Dn;

    float* ws  = (float*)d_ws;
    float* h   = ws;
    float* x   = ws + 1 * MD;
    float* q   = ws + 2 * MD;
    float* k   = ws + 3 * MD;
    float* v   = ws + 4 * MD;
    float* av  = ws + 5 * MD;
    float* t   = ws + 6 * MD;
    float* t2  = ws + 7 * MD;
    float* agg = ws + 8 * MD;
    float* sc  = ws + 9 * MD;                   // B*H*S*S fp32 (L2-resident)

    dim3 blk(TPB);
    const long long DD = (long long)Dn * Dn;
    const long long hStr  = (long long)Sn * Dn; // per-batch activation stride
    const long long scStr = (long long)Sn * Sn;

    embed_kernel<<<(unsigned)((MD + TPB - 1) / TPB), blk, 0, stream>>>(
        ids, tok, pos, h, Sn, Dn, MD);

    for (int l = 0; l < Ln; ++l) {
        const float* Wq_l = Wq + l * DD;     const float* bq_l = bq + l * Dn;
        const float* Wk_l = Wk + l * DD;     const float* bk_l = bk + l * Dn;
        const float* Wv_l = Wv + l * DD;     const float* bv_l = bv + l * Dn;
        const float* Wo_l = Wo + l * DD;     const float* bo_l = bo + l * Dn;
        const float* Wi_l = Win + l * DD;    const float* bi_l = bin_ + l * Dn;
        const float* We_l = Wedge + (long long)l * 2 * DD;
        const float* be_l = bedge + l * Dn;
        const float* Wg_l = Wgate + l * DD;  const float* bg_l = bgate + l * Dn;

        // ---- attention ----
        ln_kernel<<<M, blk, 0, stream>>>(h, ln1w + l * Dn, ln1b + l * Dn, x, Dn);
        gemm_go<1, false, false>(x, Dn, Wq_l, Dn, bq_l, nullptr, nullptr, q, Dn,
                                 M, Dn, Dn, Sn, 1, 1, 0,0,0,0,0,0, stream);
        gemm_go<1, false, false>(x, Dn, Wk_l, Dn, bk_l, nullptr, nullptr, k, Dn,
                                 M, Dn, Dn, Sn, 1, 1, 0,0,0,0,0,0, stream);
        gemm_go<1, false, false>(x, Dn, Wv_l, Dn, bv_l, nullptr, nullptr, v, Dn,
                                 M, Dn, Dn, Sn, 1, 1, 0,0,0,0,0,0, stream);

        // scores[z] = Q_z @ K_z^T,  z = b*H + h
        gemm_go<0, true, false>(q, Dn, k, Dn, nullptr, nullptr, nullptr, sc, Sn,
                                Sn, Sn, DHn, Sn, Bn * Hn, Hn,
                                hStr, DHn, hStr, DHn,
                                (long long)Hn * scStr, scStr, stream);

        softmax_causal_kernel<<<Bn * Hn * Sn, blk, 0, stream>>>(sc, Sn, 0.125f);

        // av[z] = P_z @ V_z
        gemm_go<0, false, false>(sc, Sn, v, Dn, nullptr, nullptr, nullptr, av, Dn,
                                 Sn, DHn, Sn, Sn, Bn * Hn, Hn,
                                 (long long)Hn * scStr, scStr,
                                 hStr, DHn, hStr, DHn, stream);

        // h = h + av @ Wo + bo   (each element read+written by one thread)
        gemm_go<2, false, false>(av, Dn, Wo_l, Dn, bo_l, h, nullptr, h, Dn,
                                 M, Dn, Dn, Sn, 1, 1, 0,0,0,0,0,0, stream);

        // ---- TreeFFN ----
        ln_kernel<<<M, blk, 0, stream>>>(h, ln2w + l * Dn, ln2b + l * Dn, x, Dn);
        gemm_go<1, false, false>(x, Dn, Wi_l, Dn, bi_l, nullptr, nullptr, t, Dn,
                                 M, Dn, Dn, Sn, 1, 1, 0,0,0,0,0,0, stream);

        float* tc = t;
        float* tn = t2;
        for (int it = 0; it < 2; ++it) {
            zero_first_rows_kernel<<<Bn, blk, 0, stream>>>(agg, Sn, Dn);
            // agg[b, i+1] = relu( [t[b,i], t[b,i+1]] @ Wedge + bedge )
            gemm_go<3, false, true>(tc, Dn, We_l, Dn, be_l, nullptr, nullptr,
                                    agg, Dn, Bn * (Sn - 1), Dn, 2 * Dn,
                                    Sn, 1, 1, 0,0,0,0,0,0, stream);
            // t' = t + sigmoid(agg @ Wgate + bgate) * agg
            gemm_go<4, false, false>(agg, Dn, Wg_l, Dn, bg_l, tc, agg, tn, Dn,
                                     M, Dn, Dn, Sn, 1, 1, 0,0,0,0,0,0, stream);
            float* tmp = tc; tc = tn; tn = tmp;
        }
        add_inplace_kernel<<<(unsigned)((MD + TPB - 1) / TPB), blk, 0, stream>>>(
            h, tc, MD);
    }

    // ---- final LN + LM head (dominant GEMM: 4096 x 32000 x 512) ----
    ln_kernel<<<M, blk, 0, stream>>>(h, lnfw, lnfb, x, Dn);
    gemm_go<0, false, false>(x, Dn, Whead, Vn, nullptr, nullptr, nullptr, out, Vn,
                             M, Vn, Dn, Sn, 1, 1, 0,0,0,0,0,0, stream);
}
